// vgae_decoder_58841051955354
// MI455X (gfx1250) — compile-verified, matrix operand order checked by reading, not run on hardware
//
#include <hip/hip_runtime.h>
#include <hip/hip_bf16.h>

typedef __attribute__((ext_vector_type(16))) _Float16 v16h;
typedef __attribute__((ext_vector_type(8)))  float    v8f;
typedef __attribute__((ext_vector_type(4)))  float    v4f;
typedef __attribute__((ext_vector_type(2)))  float    v2f;

#define NUM_USER 500000
#define NUM_ITEM 500000
#define DIM_E    64
#define BATCH    16384
#define NROWS    (NUM_USER + NUM_ITEM)

#define KL_BLOCKS  2048
#define KL_THREADS 256
#define DEC_BLOCKS 512      // 4 waves/block * 16 rows/wave = 32768 = 2*BATCH rows
#define WS_KL  0
#define WS_REC KL_BLOCKS

// ---------------------------------------------------------------------------
// Kernel 1: KL reduction. Pure bandwidth: streams x_mean, x_std (512 MB total)
// with nontemporal b128 loads; per-block partial -> ws (deterministic, no
// float atomics).
// ---------------------------------------------------------------------------
__global__ __launch_bounds__(KL_THREADS) void kl_kernel(const float* __restrict__ xm,
                                                        const float* __restrict__ xs,
                                                        float* __restrict__ kl_partial) {
    const int nvec = NROWS * DIM_E / 4;   // 16,000,000 float4 per array
    const v4f* m4 = (const v4f*)xm;
    const v4f* s4 = (const v4f*)xs;
    float acc = 0.0f;
    const int stride = gridDim.x * blockDim.x;
    for (int i = blockIdx.x * blockDim.x + threadIdx.x; i < nvec; i += stride) {
        v4f m = __builtin_nontemporal_load(&m4[i]);
        v4f s = __builtin_nontemporal_load(&s4[i]);
#pragma unroll
        for (int c = 0; c < 4; ++c) {
            float mu = m[c], sg = s[c];
            acc += 1.0f - mu * mu - sg * sg + 2.0f * __logf(sg);
        }
    }
    // wave32 reduction
#pragma unroll
    for (int off = 16; off > 0; off >>= 1) acc += __shfl_xor(acc, off, 32);
    __shared__ float lred[KL_THREADS / 32];
    const int lane = threadIdx.x & 31, wv = threadIdx.x >> 5;
    if (lane == 0) lred[wv] = acc;
    __syncthreads();
    if (threadIdx.x == 0) {
        float t = 0.0f;
#pragma unroll
        for (int w = 0; w < KL_THREADS / 32; ++w) t += lred[w];
        kl_partial[blockIdx.x] = t;
    }
}

// K index held in half-slot c (0..15) of a v16h operand, per CDNA5 16-bit
// A-matrix layout (half-wave selected by hsel = lane>>4).
__device__ __forceinline__ int kmap(int c, int hsel) {
    return (c < 8) ? (8 * hsel + c) : (16 + 8 * hsel + (c - 8));
}

// ---------------------------------------------------------------------------
// Kernel 2: decoder. One wave = 16 rows of either pos (s<B) or neg scores.
// h0 = relu(u*item) staged as f16 in LDS; h1 = relu(h0 @ W1 + b1) via
// v_wmma_f32_16x16x32_f16 (4 N-tiles x 2 K-chunks); h2 = sigmoid(h1 @ W2 + b2)
// via per-lane dot + 16-lane shfl reduction. Per-block sum of (score-tgt)^2
// written to ws (deterministic).
// ---------------------------------------------------------------------------
__global__ __launch_bounds__(128) void decode_kernel(const float* __restrict__ x,
                                                     const int* __restrict__ users,
                                                     const int* __restrict__ items,
                                                     const int* __restrict__ neg_items,
                                                     const float* __restrict__ W1,
                                                     const float* __restrict__ b1,
                                                     const float* __restrict__ W2f,
                                                     const float* __restrict__ b2,
                                                     float* __restrict__ rec_partial) {
    __shared__ _Float16 h0s[4][16][DIM_E];   // per-wave staging, 8 KB
    __shared__ float wred[4];

    const int lane = threadIdx.x & 31;
    const int wv   = threadIdx.x >> 5;
    const int gw   = blockIdx.x * 4 + wv;    // global wave id, 0..2047
    const int s0   = gw * 16;                // first score row of this wave
    const bool is_pos = (s0 < BATCH);
    const int  boff   = is_pos ? s0 : (s0 - BATCH);
    const int* it_idx = is_pos ? items : neg_items;

    // ---- stage h0 = relu(u * item) as f16 in LDS (coalesced v2f loads) ----
    for (int r = 0; r < 16; ++r) {
        const int ui = users[boff + r];
        const int ii = it_idx[boff + r] + NUM_USER;
        v2f uu = *(const v2f*)(x + ui * DIM_E + 2 * lane);
        v2f vv = *(const v2f*)(x + ii * DIM_E + 2 * lane);
        float p0 = uu[0] * vv[0];  p0 = p0 > 0.0f ? p0 : 0.0f;
        float p1 = uu[1] * vv[1];  p1 = p1 > 0.0f ? p1 : 0.0f;
        union { unsigned u; _Float16 h[2]; } pk;
        pk.h[0] = (_Float16)p0;  pk.h[1] = (_Float16)p1;
        *(unsigned*)&h0s[wv][r][2 * lane] = pk.u;
    }
    __syncthreads();

    const int nloc = lane & 15;   // column within 16-wide tile / row M for A
    const int hsel = lane >> 4;

    // ---- B operand: W1 as f16, 4 N-tiles x 2 K-chunks ----
    v16h Bm[4][2];
#pragma unroll
    for (int j = 0; j < 4; ++j) {
#pragma unroll
        for (int kk = 0; kk < 2; ++kk) {
            union { v16h v; _Float16 h[16]; } t;
#pragma unroll
            for (int c = 0; c < 16; ++c) {
                const int K = kmap(c, hsel) + 32 * kk;
                t.h[c] = (_Float16)W1[K * DIM_E + 16 * j + nloc];
            }
            Bm[j][kk] = t.v;
        }
    }

    // ---- WMMA accumulation: h1 tiles (16x16 f32 each) ----
    v8f ct[4] = {};
#pragma unroll
    for (int kk = 0; kk < 2; ++kk) {
        union { v16h v; unsigned u[8]; } a;
#pragma unroll
        for (int i = 0; i < 8; ++i) {
            const int K = kmap(2 * i, hsel) + 32 * kk;   // pair (K, K+1), K even
            a.u[i] = *(const unsigned*)&h0s[wv][nloc][K];
        }
#pragma unroll
        for (int j = 0; j < 4; ++j)
            ct[j] = __builtin_amdgcn_wmma_f32_16x16x32_f16(
                false, a.v, false, Bm[j][kk], (short)0, ct[j], false, false);
    }

    // ---- bias + relu + 64->1 projection ----
    // C layout: VGPR p holds row M = p + 8*hsel, column N = 16*j + nloc.
    float dot[8] = {};
#pragma unroll
    for (int j = 0; j < 4; ++j) {
        const float bj = b1[16 * j + nloc];
        const float wj = W2f[16 * j + nloc];
#pragma unroll
        for (int p = 0; p < 8; ++p) {
            float h = ct[j][p] + bj;
            h = h > 0.0f ? h : 0.0f;
            dot[p] = fmaf(h, wj, dot[p]);
        }
    }
    // reduce across the 16 lanes sharing hsel (columns)
#pragma unroll
    for (int off = 1; off < 16; off <<= 1) {
#pragma unroll
        for (int p = 0; p < 8; ++p) dot[p] += __shfl_xor(dot[p], off, 32);
    }

    const float bb  = b2[0];
    const float tgt = is_pos ? 1.0f : 0.0f;
    float local = 0.0f;
#pragma unroll
    for (int p = 0; p < 8; ++p) {
        const float z  = dot[p] + bb;
        const float sc = 1.0f / (1.0f + __expf(-z));
        const float d  = sc - tgt;
        local += d * d;
    }
    // lanes 0 (rows 0..7) and 16 (rows 8..15) hold valid group sums
    float wsum = ((lane & 15) == 0) ? local : 0.0f;
    wsum += __shfl_xor(wsum, 16, 32);
    if (lane == 0) wred[wv] = wsum;
    __syncthreads();
    if (threadIdx.x == 0)
        rec_partial[blockIdx.x] = wred[0] + wred[1] + wred[2] + wred[3];
}

// ---------------------------------------------------------------------------
// Kernel 3: deterministic finalize. loss = -0.5*klsum/NROWS + recsum/BATCH.
// ---------------------------------------------------------------------------
__global__ __launch_bounds__(256) void finalize_kernel(const float* __restrict__ ws,
                                                       float* __restrict__ out) {
    __shared__ float red[256];
    float a = 0.0f, r = 0.0f;
    for (int i = threadIdx.x; i < KL_BLOCKS;  i += 256) a += ws[WS_KL + i];
    for (int i = threadIdx.x; i < DEC_BLOCKS; i += 256) r += ws[WS_REC + i];
    red[threadIdx.x] = (-0.5f / (float)NROWS) * a + (1.0f / (float)BATCH) * r;
    __syncthreads();
    for (int s = 128; s > 0; s >>= 1) {
        if (threadIdx.x < s) red[threadIdx.x] += red[threadIdx.x + s];
        __syncthreads();
    }
    if (threadIdx.x == 0) out[0] = red[0];
}

extern "C" void kernel_launch(void* const* d_in, const int* in_sizes, int n_in,
                              void* d_out, int out_size, void* d_ws, size_t ws_size,
                              hipStream_t stream) {
    const float* x      = (const float*)d_in[0];
    const float* x_mean = (const float*)d_in[1];
    const float* x_std  = (const float*)d_in[2];
    const int*   users  = (const int*)d_in[3];
    const int*   items  = (const int*)d_in[4];
    const int*   negs   = (const int*)d_in[5];
    const float* W1     = (const float*)d_in[6];
    const float* b1     = (const float*)d_in[7];
    const float* W2     = (const float*)d_in[8];
    const float* b2     = (const float*)d_in[9];
    float* ws = (float*)d_ws;

    kl_kernel<<<KL_BLOCKS, KL_THREADS, 0, stream>>>(x_mean, x_std, ws + WS_KL);
    decode_kernel<<<DEC_BLOCKS, 128, 0, stream>>>(x, users, items, negs,
                                                  W1, b1, W2, b2, ws + WS_REC);
    finalize_kernel<<<1, 256, 0, stream>>>(ws, (float*)d_out);
}
